// CrossScaleTokenAttentionMH_55353538511610
// MI455X (gfx1250) — matrix operand order repaired and checked by reading, not compiled, not run on hardware
//
#include <hip/hip_runtime.h>

typedef __attribute__((ext_vector_type(16))) __bf16 v16bf;
typedef __attribute__((ext_vector_type(8)))  __bf16 v8bf;
typedef __attribute__((ext_vector_type(8)))  float  v8f;

#define C_DIM   256
#define N_SP    32768
#define NT      64
#define G       8
#define HEADS   4
#define DHEAD   64

__device__ __forceinline__ __bf16 f2bf(float x) { return (__bf16)x; }

// LDS activation tile layout: [n][c], XOR-swizzled in 16B (8-elem) blocks so
// per-column K-runs are contiguous (ds_load_b128) while banks stay spread.
__device__ __forceinline__ int actIdx(int n, int c) {
    return n * C_DIM + (c ^ ((n & 7) << 3));
}

__device__ __forceinline__ v16bf frag16(v8bf lo, v8bf hi) {
    return __builtin_shufflevector(lo, hi, 0, 1, 2, 3, 4, 5, 6, 7,
                                           8, 9, 10, 11, 12, 13, 14, 15);
}

// ---------------------------------------------------------------------------
// Kernel 0: convert q_w / proj_w fp32 [o][c] into fragment-ready swizzled bf16.
// Output index: (((mtile*8)+ktile)*32 + lane)*16 + e  holds
//   W[mtile*16 + (lane&15)][ktile*32 + e + ((lane<16)?0:16)]
// so each lane's A fragment is one contiguous 32B run (2x global_load_b128).
// ---------------------------------------------------------------------------
__global__ void convert_weights(const float* __restrict__ qw, const float* __restrict__ pw,
                                __bf16* __restrict__ qwb, __bf16* __restrict__ pwb) {
    int idx  = blockIdx.x * blockDim.x + threadIdx.x;   // 0..65535
    int e    = idx & 15;
    int lane = (idx >> 4) & 31;
    int kt   = (idx >> 9) & 7;
    int mt   = idx >> 12;
    int row  = mt * 16 + (lane & 15);
    int k    = kt * 32 + e + ((lane < 16) ? 0 : 16);
    qwb[idx] = f2bf(qw[row * C_DIM + k]);
    pwb[idx] = f2bf(pw[row * C_DIM + k]);
}

// ---------------------------------------------------------------------------
// Kernel 1: adaptive_avg_pool3d 16^3 -> 2^3 (8x8x8 block means)
// ---------------------------------------------------------------------------
__global__ void pool_kernel(const float* __restrict__ x1, float* __restrict__ tokens_raw) {
    const int bc   = blockIdx.x;
    const int gp   = threadIdx.x >> 5;
    const int lane = threadIdx.x & 31;
    const int gz = gp >> 2, gy = (gp >> 1) & 1, gx = gp & 1;
    const float* base = x1 + (size_t)bc * 4096;
    float s = 0.f;
#pragma unroll
    for (int i = 0; i < 16; ++i) {
        int e  = lane + 32 * i;
        int dz = e >> 6, dy = (e >> 3) & 7, dx = e & 7;
        s += base[((gz * 8 + dz) * 16 + (gy * 8 + dy)) * 16 + (gx * 8 + dx)];
    }
#pragma unroll
    for (int off = 16; off; off >>= 1) s += __shfl_xor(s, off, 32);
    if (lane == 0) tokens_raw[(size_t)bc * G + gp] = s * (1.0f / 512.0f);
}

// ---------------------------------------------------------------------------
// Kernel 2: depthwise 3x3x3 on 2x2x2 tokens + channel-LN + K/V projections.
// K stored [b][h][d][g] with 1/sqrt(d) folded in; V stored [b][h][g][d].
// ---------------------------------------------------------------------------
__global__ void tokens_kv_kernel(const float* __restrict__ tokens_raw,
                                 const float* __restrict__ dw_w,
                                 const float* __restrict__ k_w,
                                 const float* __restrict__ v_w,
                                 const float* __restrict__ nkv_w,
                                 const float* __restrict__ nkv_b,
                                 float* __restrict__ k_ws,
                                 float* __restrict__ v_ws) {
    __shared__ float tok[C_DIM * G];
    __shared__ float tkk[C_DIM * G];
    __shared__ float mu[G], rs[G];
    const int b = blockIdx.x;
    const int c = threadIdx.x;

    float t0[G];
    const float* tr = tokens_raw + ((size_t)b * C_DIM + c) * G;
#pragma unroll
    for (int g = 0; g < G; ++g) t0[g] = tr[g];

    const float* w = dw_w + c * 27;
    float tf[G];
#pragma unroll
    for (int z = 0; z < 2; ++z)
#pragma unroll
    for (int y = 0; y < 2; ++y)
#pragma unroll
    for (int x = 0; x < 2; ++x) {
        float acc = 0.f;
#pragma unroll
        for (int kz = 0; kz < 3; ++kz) {
            int zi = z + kz - 1; if (zi < 0 || zi > 1) continue;
#pragma unroll
            for (int ky = 0; ky < 3; ++ky) {
                int yi = y + ky - 1; if (yi < 0 || yi > 1) continue;
#pragma unroll
                for (int kx = 0; kx < 3; ++kx) {
                    int xi = x + kx - 1; if (xi < 0 || xi > 1) continue;
                    acc += t0[zi * 4 + yi * 2 + xi] * w[kz * 9 + ky * 3 + kx];
                }
            }
        }
        tf[z * 4 + y * 2 + x] = t0[z * 4 + y * 2 + x] + acc;
    }
#pragma unroll
    for (int g = 0; g < G; ++g) tok[c * G + g] = tf[g];
    __syncthreads();

    if (c < G) {
        float s = 0.f, s2 = 0.f;
        for (int cc = 0; cc < C_DIM; ++cc) { float v = tok[cc * G + c]; s += v; s2 += v * v; }
        float m = s * (1.0f / 256.0f);
        mu[c] = m;
        rs[c] = rsqrtf(s2 * (1.0f / 256.0f) - m * m + 1e-6f);
    }
    __syncthreads();

    const float wl = nkv_w[c], bl = nkv_b[c];
#pragma unroll
    for (int g = 0; g < G; ++g) tkk[c * G + g] = (tf[g] - mu[g]) * rs[g] * wl + bl;
    __syncthreads();

    const float* krow = k_w + (size_t)c * C_DIM;
    const float* vrow = v_w + (size_t)c * C_DIM;
    float kacc[G], vacc[G];
#pragma unroll
    for (int g = 0; g < G; ++g) { kacc[g] = 0.f; vacc[g] = 0.f; }
    for (int cc = 0; cc < C_DIM; ++cc) {
        float kw = krow[cc], vw = vrow[cc];
#pragma unroll
        for (int g = 0; g < G; ++g) {
            kacc[g] += kw * tkk[cc * G + g];
            vacc[g] += vw * tok[cc * G + g];
        }
    }
    const int h = c >> 6, d = c & 63;
#pragma unroll
    for (int g = 0; g < G; ++g) {
        k_ws[((size_t)b * C_DIM + c) * G + g]         = kacc[g] * 0.125f;  // 1/sqrt(64)
        v_ws[(size_t)b * 2048 + h * 512 + g * 64 + d] = vacc[g];
    }
}

// ---------------------------------------------------------------------------
// Kernel 3: fused LN(x2) -> q-GEMM (WMMA bf16) -> 8-token attention -> proj-GEMM
// grid = (N/64, B), block = 256 (8 waves); wave owns rows [w*32, w*32+32) x 64 cols.
// ---------------------------------------------------------------------------
__global__ void __launch_bounds__(256)
main_attn_kernel(const float* __restrict__ x2,
                 const __bf16* __restrict__ qwb,
                 const __bf16* __restrict__ pwb,
                 const float* __restrict__ k_ws,
                 const float* __restrict__ v_ws,
                 const float* __restrict__ nq_w,
                 const float* __restrict__ nq_b,
                 float* __restrict__ out) {
    __shared__ __align__(16) unsigned char smem[65536];
    __bf16* buf0    = (__bf16*)smem;            // 64x256 bf16 [n][c]: qx, then q
    __bf16* buf1    = (__bf16*)(smem + 32768);  // 64x256 bf16 [n][c]: attn output
    float*  scratch = (float*)(smem + 32768);   // LN scratch (dead before buf1 use)

    const int b  = blockIdx.y;
    const int n0 = blockIdx.x * NT;
    const int t  = threadIdx.x;
    const int cl = t >> 6, j = t & 63;          // channel quarter / column

    // ---- channel LayerNorm pass 1: partial sums over 64 channels each ----
    {
        const float* xp = x2 + (size_t)b * C_DIM * N_SP + n0 + j;
        float s = 0.f, s2 = 0.f;
#pragma unroll 4
        for (int i = 0; i < 64; ++i) {
            float v = xp[(size_t)(cl * 64 + i) * N_SP];
            s += v; s2 += v * v;
        }
        scratch[cl * 64 + j]       = s;
        scratch[256 + cl * 64 + j] = s2;
    }
    __syncthreads();
    if (t < 64) {
        float s  = scratch[t] + scratch[64 + t] + scratch[128 + t] + scratch[192 + t];
        float s2 = scratch[256 + t] + scratch[320 + t] + scratch[384 + t] + scratch[448 + t];
        float m  = s * (1.0f / 256.0f);
        scratch[512 + t] = m;
        scratch[576 + t] = rsqrtf(s2 * (1.0f / 256.0f) - m * m + 1e-6f);
    }
    __syncthreads();
    // ---- pass 2: normalize + affine, write bf16 qx into LDS (16B stores) ----
    {
        const float m = scratch[512 + j];
        const float r = scratch[576 + j];
        const float* xp = x2 + (size_t)b * C_DIM * N_SP + n0 + j;
#pragma unroll 1
        for (int ib = 0; ib < 8; ++ib) {
            v8bf qv;
#pragma unroll
            for (int u = 0; u < 8; ++u) {
                int c = cl * 64 + ib * 8 + u;
                float v = xp[(size_t)c * N_SP];
                qv[u] = f2bf((v - m) * r * nq_w[c] + nq_b[c]);
            }
            *(v8bf*)(buf0 + actIdx(j, cl * 64 + ib * 8)) = qv;
        }
    }
    __syncthreads();

    // ---- WMMA coordinates (16x16x32 bf16, wave32) ----
    const int wave  = t >> 5;
    const int lane  = t & 31;
    const int m0    = wave * 32;
    const int cB    = lane & 15;
    const int kBoff = (lane < 16) ? 0 : 16;
    const int rowHi = (lane < 16) ? 0 : 8;

    v8f acc[2][4];
#pragma unroll
    for (int mt = 0; mt < 2; ++mt)
#pragma unroll
        for (int nt = 0; nt < 4; ++nt)
#pragma unroll
            for (int e = 0; e < 8; ++e) acc[mt][nt][e] = 0.f;

    // ---- GEMM1: q = q_w x qx ----
    for (int k0 = 0; k0 < 256; k0 += 32) {
        const int kt = k0 >> 5;
        const __bf16* a0p = qwb + (((size_t)(wave * 2)     * 8 + kt) * 32 + lane) * 16;
        const __bf16* a1p = qwb + (((size_t)(wave * 2 + 1) * 8 + kt) * 32 + lane) * 16;
        v16bf a0 = frag16(*(const v8bf*)a0p, *(const v8bf*)(a0p + 8));
        v16bf a1 = frag16(*(const v8bf*)a1p, *(const v8bf*)(a1p + 8));
        const int kb = k0 + kBoff;
#pragma unroll
        for (int nt = 0; nt < 4; ++nt) {
            const int ncol = nt * 16 + cB;
            v16bf bf = frag16(*(const v8bf*)(buf0 + actIdx(ncol, kb)),
                              *(const v8bf*)(buf0 + actIdx(ncol, kb + 8)));
            acc[0][nt] = __builtin_amdgcn_wmma_f32_16x16x32_bf16(false, a0, false, bf,
                                                                 (short)0, acc[0][nt], false, false);
            acc[1][nt] = __builtin_amdgcn_wmma_f32_16x16x32_bf16(false, a1, false, bf,
                                                                 (short)0, acc[1][nt], false, false);
        }
    }
    __syncthreads();           // all qx reads complete -> reuse buf0 for q
#pragma unroll
    for (int mt = 0; mt < 2; ++mt)
#pragma unroll
        for (int nt = 0; nt < 4; ++nt) {
            const int col   = nt * 16 + cB;
            const int mbase = m0 + mt * 16 + rowHi;   // multiple of 8
            v8bf qv;
#pragma unroll
            for (int r = 0; r < 8; ++r) qv[r] = f2bf(acc[mt][nt][r]);
            *(v8bf*)(buf0 + actIdx(col, mbase)) = qv;
        }
    __syncthreads();

    // ---- attention: thread = (head h, column j); 8 tokens ----
    {
        const int h = t >> 6;
        const float* kp = k_ws + ((size_t)b * C_DIM + h * DHEAD) * G;  // [d][g], pre-scaled
        const float* vp = v_ws + (size_t)b * 2048 + h * 512;           // [g][d]
        float l[G];
#pragma unroll
        for (int g = 0; g < G; ++g) l[g] = 0.f;
#pragma unroll
        for (int db = 0; db < 8; ++db) {
            v8bf qv = *(const v8bf*)(buf0 + actIdx(j, h * DHEAD + db * 8));
#pragma unroll
            for (int r = 0; r < 8; ++r) {
                float q = (float)qv[r];
                const float* kd = kp + (db * 8 + r) * G;
#pragma unroll
                for (int g = 0; g < G; ++g) l[g] += q * kd[g];
            }
        }
        float mx = l[0];
#pragma unroll
        for (int g = 1; g < G; ++g) mx = fmaxf(mx, l[g]);
        float sum = 0.f;
#pragma unroll
        for (int g = 0; g < G; ++g) { l[g] = __expf(l[g] - mx); sum += l[g]; }
        float inv = 1.0f / sum;
#pragma unroll
        for (int g = 0; g < G; ++g) l[g] *= inv;
#pragma unroll
        for (int db = 0; db < 8; ++db) {
            float o[8];
#pragma unroll
            for (int r = 0; r < 8; ++r) o[r] = 0.f;
#pragma unroll
            for (int g = 0; g < G; ++g) {
                float ag = l[g];
                const float* vg = vp + g * DHEAD + db * 8;
#pragma unroll
                for (int r = 0; r < 8; ++r) o[r] += ag * vg[r];
            }
            v8bf ov;
#pragma unroll
            for (int r = 0; r < 8; ++r) ov[r] = f2bf(o[r]);
            *(v8bf*)(buf1 + actIdx(j, h * DHEAD + db * 8)) = ov;
        }
    }
    __syncthreads();

    // ---- GEMM2: final = proj_w x attn_out ----
#pragma unroll
    for (int mt = 0; mt < 2; ++mt)
#pragma unroll
        for (int nt = 0; nt < 4; ++nt)
#pragma unroll
            for (int e = 0; e < 8; ++e) acc[mt][nt][e] = 0.f;

    for (int k0 = 0; k0 < 256; k0 += 32) {
        const int kt = k0 >> 5;
        const __bf16* a0p = pwb + (((size_t)(wave * 2)     * 8 + kt) * 32 + lane) * 16;
        const __bf16* a1p = pwb + (((size_t)(wave * 2 + 1) * 8 + kt) * 32 + lane) * 16;
        v16bf a0 = frag16(*(const v8bf*)a0p, *(const v8bf*)(a0p + 8));
        v16bf a1 = frag16(*(const v8bf*)a1p, *(const v8bf*)(a1p + 8));
        const int kb = k0 + kBoff;
#pragma unroll
        for (int nt = 0; nt < 4; ++nt) {
            const int ncol = nt * 16 + cB;
            v16bf bf = frag16(*(const v8bf*)(buf1 + actIdx(ncol, kb)),
                              *(const v8bf*)(buf1 + actIdx(ncol, kb + 8)));
            acc[0][nt] = __builtin_amdgcn_wmma_f32_16x16x32_bf16(false, a0, false, bf,
                                                                 (short)0, acc[0][nt], false, false);
            acc[1][nt] = __builtin_amdgcn_wmma_f32_16x16x32_bf16(false, a1, false, bf,
                                                                 (short)0, acc[1][nt], false, false);
        }
    }

    // ---- store f32 result per C/D layout (coalesced dword stores) ----
    float* op = out + (size_t)b * C_DIM * N_SP + n0;
#pragma unroll
    for (int mt = 0; mt < 2; ++mt)
#pragma unroll
        for (int nt = 0; nt < 4; ++nt)
#pragma unroll
            for (int r = 0; r < 8; ++r)
                op[(size_t)(m0 + mt * 16 + rowHi + r) * N_SP + nt * 16 + cB] = acc[mt][nt][r];
}

// ---------------------------------------------------------------------------
extern "C" void kernel_launch(void* const* d_in, const int* in_sizes, int n_in,
                              void* d_out, int out_size, void* d_ws, size_t ws_size,
                              hipStream_t stream) {
    const float* x2     = (const float*)d_in[0];
    const float* x1_low = (const float*)d_in[1];
    const float* q_w    = (const float*)d_in[2];
    const float* k_w    = (const float*)d_in[3];
    const float* v_w    = (const float*)d_in[4];
    const float* dw_w   = (const float*)d_in[5];
    const float* proj_w = (const float*)d_in[6];
    const float* nq_w   = (const float*)d_in[7];
    const float* nq_b   = (const float*)d_in[8];
    const float* nkv_w  = (const float*)d_in[9];
    const float* nkv_b  = (const float*)d_in[10];
    float* out = (float*)d_out;

    char* ws = (char*)d_ws;
    __bf16* qwb        = (__bf16*)ws;              // 65536 bf16 (swizzled)
    __bf16* pwb        = qwb + 65536;              // 65536 bf16 (swizzled)
    float*  tokens_raw = (float*)(ws + 262144);    // 2*256*8
    float*  k_ws       = tokens_raw + 4096;        // 2*256*8
    float*  v_ws       = k_ws + 4096;              // 2*256*8

    convert_weights<<<256, 256, 0, stream>>>(q_w, proj_w, qwb, pwb);
    pool_kernel<<<512, 256, 0, stream>>>(x1_low, tokens_raw);
    tokens_kv_kernel<<<2, 256, 0, stream>>>(tokens_raw, dw_w, k_w, v_w, nkv_w, nkv_b, k_ws, v_ws);
    dim3 grid(N_SP / NT, 2);
    main_attn_kernel<<<grid, 256, 0, stream>>>(x2, qwb, pwb, k_ws, v_ws, nq_w, nq_b, out);
}
